// DeltaNet_31877247271551
// MI455X (gfx1250) — compile-verified, matrix operand order checked by reading, not run on hardware
//
#include <hip/hip_runtime.h>

// ---------------------------------------------------------------------------
// DeltaNet hybrid layer for MI455X (gfx1250, wave32, WMMA).
// Heavy GEMMs: v_wmma_f32_16x16x32_f16 fed by a double-buffered
// GLOBAL_LOAD_ASYNC_TO_LDS_B128 pipeline (ASYNCcnt-tracked DMA, no data
// VGPRs in the fill path). All fragments are contiguous 128-bit loads.
// ---------------------------------------------------------------------------

typedef __attribute__((ext_vector_type(16))) _Float16 v16h;
typedef __attribute__((ext_vector_type(8)))  float    v8f;
typedef __attribute__((ext_vector_type(8)))  _Float16 h8;

#define NT 256           // threads per block (8 wave32s)
#define Mrows 8192       // B*L
#define Dmodel 1024
#define Hh 4
#define DKV 256
#define CHK 32
#define NCHK 64
#define LSEQ 2048

// ---------------- CDNA5 async memory->LDS helpers --------------------------

__device__ __forceinline__ void async_g2l_b128(unsigned lds_addr,
                                               const _Float16* gaddr) {
  // per-lane: LDS[lds_addr .. +15] = MEM[gaddr .. +15]; tracked by ASYNCcnt
  asm volatile("global_load_async_to_lds_b128 %0, %1, off"
               :: "v"(lds_addr), "v"(gaddr)
               : "memory");
}
__device__ __forceinline__ void s_wait_async0() {
  asm volatile("s_wait_asynccnt 0x0" ::: "memory");
}

// ---------------- WMMA fragment helpers (layouts per CDNA5 ISA 7.12.2) -----

__device__ __forceinline__ v8f wmma16(v16h a, v16h b, v8f c) {
  // 8 args: (neg_a, A, neg_b, B, c_mod, C, reuse_a, reuse_b)
  return __builtin_amdgcn_wmma_f32_16x16x32_f16(false, a, false, b,
                                                (short)0, c, false, false);
}

// A fragment: 16x32 row-major tile at p, row stride ld (contiguous K).
// lane<16: row=lane, K={0..7,16..23}; lane>=16: row=lane-16, K={8..15,24..31}
__device__ __forceinline__ v16h load_a(const _Float16* p, int ld) {
  int lane = threadIdx.x & 31;
  int row = lane & 15, ko = (lane >> 4) * 8;
  const _Float16* r = p + row * ld + ko;
  h8 lo = *(const h8*)&r[0];
  h8 hi = *(const h8*)&r[16];
  v16h a;
#pragma unroll
  for (int j = 0; j < 8; ++j) { a[j] = lo[j]; a[8 + j] = hi[j]; }
  return a;
}

// B fragment where logical B = X^T with X row-major (16 x 32 tile at p, ld):
// B[k][n] = X[n][k]  -> contiguous per lane (two b128 loads).
__device__ __forceinline__ v16h load_bt(const _Float16* p, int ld) {
  int lane = threadIdx.x & 31;
  int n = lane & 15, ko = (lane >> 4) * 8;
  const _Float16* r = p + n * ld + ko;
  h8 lo = *(const h8*)&r[0];
  h8 hi = *(const h8*)&r[16];
  v16h b;
#pragma unroll
  for (int j = 0; j < 8; ++j) { b[j] = lo[j]; b[8 + j] = hi[j]; }
  return b;
}

// C/D 16x16 f32: VGPR r: lanes0-15 (M=r,N=lane), lanes16-31 (M=r+8,N=lane-16)
__device__ __forceinline__ void store_cf(float* p, int ld, v8f c) {
  int lane = threadIdx.x & 31;
  int n = lane & 15, mo = (lane >> 4) * 8;
#pragma unroll
  for (int r = 0; r < 8; ++r) p[(mo + r) * ld + n] = c[r];
}
__device__ __forceinline__ void store_ch(_Float16* p, int ld, v8f c, float sc) {
  int lane = threadIdx.x & 31;
  int n = lane & 15, mo = (lane >> 4) * 8;
#pragma unroll
  for (int r = 0; r < 8; ++r) p[(mo + r) * ld + n] = (_Float16)(c[r] * sc);
}
__device__ __forceinline__ v8f load_ch(const _Float16* p, int ld) {
  int lane = threadIdx.x & 31;
  int n = lane & 15, mo = (lane >> 4) * 8;
  v8f c;
#pragma unroll
  for (int r = 0; r < 8; ++r) c[r] = (float)p[(mo + r) * ld + n];
  return c;
}
// transposed f16 store of a C tile: dst[n][m] layout -> per-lane CONTIGUOUS 16B
__device__ __forceinline__ void store_chT(_Float16* p, int ld, v8f c) {
  int lane = threadIdx.x & 31;
  int n = lane & 15, mo = (lane >> 4) * 8;
  h8 t;
#pragma unroll
  for (int r = 0; r < 8; ++r) t[r] = (_Float16)c[r];
  *(h8*)&p[n * ld + mo] = t;
}

// ---------------- generic tiled WMMA GEMM (async double-buffered) ----------
// C = A(MxK,f16) * Bt^T, Bt is N x K (row-major) pre-transposed weight;
// optional bias; f32 out. M%128==0, N%128==0, K%32==0.
__global__ __launch_bounds__(NT) void wmma_gemm(
    const _Float16* __restrict__ A, const _Float16* __restrict__ Bt,
    const float* __restrict__ bias, float* __restrict__ C,
    int M, int N, int Kd) {
  __shared__ _Float16 sA[2][128][32];
  __shared__ _Float16 sBt[2][128][32];
  int tid = threadIdx.x, lane = tid & 31, wid = tid >> 5;
  int bm = blockIdx.x * 128, bn = blockIdx.y * 128;
  int wm = wid >> 1, wn = wid & 1;   // 4x2 wave grid: 32x64 per wave

  // each thread DMAs two 16B chunks of each tile (rows r and r+64)
  int r = tid >> 2, o = (tid & 3) * 8;

  v8f acc[2][4];
#pragma unroll
  for (int i = 0; i < 2; ++i)
#pragma unroll
    for (int j = 0; j < 4; ++j) acc[i][j] = (v8f){};

  int nk = Kd / 32;
  // prologue: DMA tile 0 into buffer 0
  {
    async_g2l_b128((unsigned)(size_t)&sA[0][r][o],       &A[(size_t)(bm + r) * Kd + o]);
    async_g2l_b128((unsigned)(size_t)&sA[0][r + 64][o],  &A[(size_t)(bm + r + 64) * Kd + o]);
    async_g2l_b128((unsigned)(size_t)&sBt[0][r][o],      &Bt[(size_t)(bn + r) * Kd + o]);
    async_g2l_b128((unsigned)(size_t)&sBt[0][r + 64][o], &Bt[(size_t)(bn + r + 64) * Kd + o]);
  }

  for (int i = 0; i < nk; ++i) {
    s_wait_async0();       // this wave's DMAs for tile i are in LDS
    __syncthreads();       // whole tile i visible; everyone done with buf i+1
    if (i + 1 < nk) {      // DMA tile i+1 into the other buffer
      int nb = (i + 1) & 1, kk = (i + 1) * 32;
      async_g2l_b128((unsigned)(size_t)&sA[nb][r][o],       &A[(size_t)(bm + r) * Kd + kk + o]);
      async_g2l_b128((unsigned)(size_t)&sA[nb][r + 64][o],  &A[(size_t)(bm + r + 64) * Kd + kk + o]);
      async_g2l_b128((unsigned)(size_t)&sBt[nb][r][o],      &Bt[(size_t)(bn + r) * Kd + kk + o]);
      async_g2l_b128((unsigned)(size_t)&sBt[nb][r + 64][o], &Bt[(size_t)(bn + r + 64) * Kd + kk + o]);
    }
    int cb = i & 1;
    v16h af[2], bf[4];
#pragma unroll
    for (int ii = 0; ii < 2; ++ii) af[ii] = load_a(&sA[cb][wm * 32 + ii * 16][0], 32);
#pragma unroll
    for (int j = 0; j < 4; ++j) bf[j] = load_bt(&sBt[cb][wn * 64 + j * 16][0], 32);
#pragma unroll
    for (int ii = 0; ii < 2; ++ii)
#pragma unroll
      for (int j = 0; j < 4; ++j) acc[ii][j] = wmma16(af[ii], bf[j], acc[ii][j]);
  }

  int n_l = lane & 15, mo = (lane >> 4) * 8;
#pragma unroll
  for (int i = 0; i < 2; ++i)
#pragma unroll
    for (int j = 0; j < 4; ++j) {
      int m0 = bm + wm * 32 + i * 16, n0 = bn + wn * 64 + j * 16;
      float bv = bias ? bias[n0 + n_l] : 0.f;
#pragma unroll
      for (int rr = 0; rr < 8; ++rr)
        C[(size_t)(m0 + mo + rr) * N + n0 + n_l] = acc[i][j][rr] + bv;
    }
}

// ---------------- elementwise / small kernels ------------------------------

__global__ __launch_bounds__(NT) void cvt_f16(const float* __restrict__ s,
                                              _Float16* __restrict__ d,
                                              long long n) {
  long long i = (long long)blockIdx.x * NT + threadIdx.x;
  if (i < n) d[i] = (_Float16)s[i];
}

// f32 (K x N, row-major) -> f16 TRANSPOSED (N x K, row-major)
__global__ __launch_bounds__(NT) void cvt_f16_T(const float* __restrict__ s,
                                                _Float16* __restrict__ d,
                                                long long n, int N, int Kd) {
  long long i = (long long)blockIdx.x * NT + threadIdx.x;
  if (i >= n) return;
  long long k = i / N;
  int nn = (int)(i - k * N);
  d[(size_t)nn * Kd + k] = (_Float16)s[i];
}

// pack f32 (rows x C) into f16 dst with row stride dld at column offset
__global__ __launch_bounds__(NT) void pack16(const float* __restrict__ s,
                                             _Float16* __restrict__ d,
                                             long long total, int C, int dld,
                                             int coff) {
  long long i = (long long)blockIdx.x * NT + threadIdx.x;
  if (i >= total) return;
  long long r = i / C; int c = (int)(i - r * C);
  d[r * dld + coff + c] = (_Float16)s[i];
}

// causal depthwise conv K=4 + SiLU;  X,Y: (B,L,C) f32, W: (C,4)
__global__ __launch_bounds__(NT) void dwconv4_silu(const float* __restrict__ X,
                                                   const float* __restrict__ W,
                                                   float* __restrict__ Y,
                                                   long long total, int L, int C) {
  long long i = (long long)blockIdx.x * NT + threadIdx.x;
  if (i >= total) return;
  int c = (int)(i % C);
  long long bl = i / C;
  int l = (int)(bl % L);
  long long b = bl / L;
  float s = 0.f;
#pragma unroll
  for (int t = 0; t < 4; ++t) {
    int ls = l - 3 + t;
    if (ls >= 0) s += X[(b * L + ls) * (long long)C + c] * W[c * 4 + t];
  }
  Y[i] = s / (1.f + expf(-s));   // SiLU
}

// depthwise causal FIR (no activation); also packs f16 into router input
__global__ __launch_bounds__(NT) void fir_kernel(const float* __restrict__ V,
                                                 const float* __restrict__ F,
                                                 float* __restrict__ Y,
                                                 _Float16* __restrict__ rin,
                                                 int Kf, int coff,
                                                 long long total, int L, int C) {
  long long i = (long long)blockIdx.x * NT + threadIdx.x;
  if (i >= total) return;
  int c = (int)(i % C);
  long long bl = i / C;
  int l = (int)(bl % L);
  long long b = bl / L;
  float s = 0.f;
  for (int t = 0; t < Kf; ++t) {
    int ls = l - (Kf - 1) + t;
    if (ls >= 0) s += V[(b * L + ls) * (long long)C + c] * F[c * Kf + t];
  }
  Y[i] = s;
  rin[bl * 6144 + coff + c] = (_Float16)s;
}

__global__ __launch_bounds__(NT) void gelu_kernel(float* __restrict__ x,
                                                  long long n) {
  long long i = (long long)blockIdx.x * NT + threadIdx.x;
  if (i < n) {
    float v = x[i];
    x[i] = 0.5f * v * (1.f + erff(v * 0.70710678118654752f));
  }
}

// one wave32 per output element: C[m,n] = act(dot(A[m,:],B[:,n]) + bias[n])
__global__ __launch_bounds__(NT) void dot_gemm(const float* __restrict__ A,
                                               const float* __restrict__ B,
                                               const float* __restrict__ bias,
                                               float* __restrict__ C,
                                               int M, int N, int Kd, int act) {
  int gid = blockIdx.x * 8 + (threadIdx.x >> 5);
  int lane = threadIdx.x & 31;
  if (gid >= M * N) return;
  int m = gid / N, n = gid - m * N;
  const float* a = A + (size_t)m * Kd;
  float s = 0.f;
  for (int k = lane; k < Kd; k += 32) s += a[k] * B[(size_t)k * N + n];
#pragma unroll
  for (int off = 16; off > 0; off >>= 1) s += __shfl_down(s, off, 32);
  if (lane == 0) {
    if (bias) s += bias[n];
    if (act == 1) s = 1.f / (1.f + expf(-s));
    C[(size_t)m * N + n] = s;
  }
}

// ---------------- delta-rule preparation -----------------------------------

__device__ __forceinline__ float blockReduceSum(float v, float* red) {
  int tid = threadIdx.x;
  red[tid] = v;
  __syncthreads();
  for (int s = 128; s > 0; s >>= 1) {
    if (tid < s) red[tid] += red[tid + s];
    __syncthreads();
  }
  float r = red[0];
  __syncthreads();
  return r;
}

// l2norm q,k ; v*beta ; kbeta. Row-major (B,H,L,256) f16 for A-side operands
// plus chunk-transposed (B,H,NCHK,256,32) f16 copies for B-side operands.
__global__ __launch_bounds__(NT) void delta_prep(
    const float* __restrict__ q, const float* __restrict__ k,
    const float* __restrict__ v, const float* __restrict__ beta,
    _Float16* __restrict__ qh, _Float16* __restrict__ kh,
    _Float16* __restrict__ kbh, _Float16* __restrict__ kT,
    _Float16* __restrict__ vT, _Float16* __restrict__ kbT) {
  __shared__ float red[NT];
  int idx = blockIdx.x;          // (b*L + l)*H + h
  int h = idx & 3;
  int bl = idx >> 2;
  int d = threadIdx.x;
  size_t src = (size_t)bl * Dmodel + h * DKV + d;
  float qv = q[src], kv = k[src], vv = v[src];
  float sq = blockReduceSum(qv * qv, red);
  float rq = rsqrtf(sq + 1e-6f);
  float sk = blockReduceSum(kv * kv, red);
  float rk = rsqrtf(sk + 1e-6f);
  float bt = beta[(size_t)bl * Hh + h];
  int b = bl / LSEQ, l = bl % LSEQ;
  _Float16 qn = (_Float16)(qv * rq);
  _Float16 kn = (_Float16)(kv * rk);
  _Float16 vb = (_Float16)(vv * bt);
  _Float16 kbn = (_Float16)(kv * rk * bt);
  size_t o = ((size_t)(b * Hh + h) * LSEQ + l) * DKV + d;
  size_t ot = (((size_t)(b * Hh + h) * NCHK + (l >> 5)) * DKV + d) * CHK + (l & 31);
  qh[o] = qn;
  kh[o] = kn;
  kbh[o] = kbn;
  kT[ot] = kn;
  vT[ot] = vb;
  kbT[ot] = kbn;
}

// ---------------- delta-rule chunkwise scan (WMMA) -------------------------
// One block per (b,h). State kept TRANSPOSED: St[dv][dk] (f16, workspace),
// so every state fragment is a contiguous-K b128 load.
__global__ __launch_bounds__(NT) void delta_kernel(
    const _Float16* __restrict__ qh, const _Float16* __restrict__ kh,
    const _Float16* __restrict__ kbh, const _Float16* __restrict__ kT,
    const _Float16* __restrict__ vT, const _Float16* __restrict__ kbT,
    _Float16* __restrict__ Sg, float* __restrict__ dout) {
  __shared__ float    sA[CHK][CHK];        // chunk A (f32)
  __shared__ float    sInv[CHK][CHK + 1];  // (I + tril(A,-1))^{-1}
  __shared__ _Float16 sInvH[CHK][CHK];
  __shared__ _Float16 sW[CHK][DKV];        // -w = -(inv @ kbeta_c), row-major
  __shared__ _Float16 sWT[DKV][CHK];       // u_adj TRANSPOSED: [dv][c]
  __shared__ _Float16 sAtt[CHK][CHK];      // masked attention (f16)

  int bh = blockIdx.x;                      // b*H + h
  int b = bh >> 2, h = bh & 3;
  const _Float16* Q  = qh  + (size_t)bh * LSEQ * DKV;
  const _Float16* K  = kh  + (size_t)bh * LSEQ * DKV;
  const _Float16* KB = kbh + (size_t)bh * LSEQ * DKV;
  _Float16* St = Sg + (size_t)bh * DKV * DKV;   // St[dv][dk]

  int tid = threadIdx.x, lane = tid & 31, wid = tid >> 5;
  int mt = wid >> 2;              // 0..1  (chunk-row halves / dv halves)
  int nt0 = (wid & 3) * 4;        // 4 column tiles of 16 per wave

  for (int i = tid; i < DKV * DKV; i += NT) St[i] = (_Float16)0.f;
  __threadfence();
  __syncthreads();

  for (int ch = 0; ch < NCHK; ++ch) {
    const _Float16* Qc   = Q  + (size_t)ch * CHK * DKV;
    const _Float16* Kc   = K  + (size_t)ch * CHK * DKV;
    const _Float16* KBc  = KB + (size_t)ch * CHK * DKV;
    const _Float16* kTc  = kT  + ((size_t)bh * NCHK + ch) * DKV * CHK;
    const _Float16* vTc  = vT  + ((size_t)bh * NCHK + ch) * DKV * CHK;
    const _Float16* kbTc = kbT + ((size_t)bh * NCHK + ch) * DKV * CHK;

    // step 1: A = KB_c @ K_c^T  (32x32, contraction over 256) — waves 0..3
    if (wid < 4) {
      int tm = wid & 1, tn = wid >> 1;
      v8f acc = (v8f){};
      for (int kk = 0; kk < DKV; kk += 32) {
        v16h a = load_a(KBc + (size_t)tm * 16 * DKV + kk, DKV);
        v16h bf = load_bt(Kc + (size_t)tn * 16 * DKV + kk, DKV);
        acc = wmma16(a, bf, acc);
      }
      store_cf(&sA[tm * 16][tn * 16], CHK, acc);
    }
    __syncthreads();

    // step 2: inv = (I + tril(A,-1))^{-1} — wave 0, lane j owns column j
    if (wid == 0) {
      int j = lane;
      sInv[0][j] = (j == 0) ? 1.f : 0.f;
      for (int i = 1; i < CHK; ++i) {
        float s = 0.f;
        for (int m = j; m < i; ++m) s += sA[i][m] * sInv[m][j];
        sInv[i][j] = (i == j) ? 1.f : ((j < i) ? -s : 0.f);
      }
      for (int i = 0; i < CHK; ++i) sInvH[i][j] = (_Float16)sInv[i][j];
    }
    __syncthreads();

    // step 3: u (registers) = inv @ V_c ;  sW = -(inv @ KB_c)
    v8f ureg[4];
    {
      v16h a = load_a(&sInvH[mt * 16][0], CHK);
#pragma unroll
      for (int t = 0; t < 4; ++t) {
        int nt = nt0 + t;
        v16h bv = load_bt(vTc + (size_t)(nt * 16) * CHK, CHK);
        ureg[t] = wmma16(a, bv, (v8f){});
        v16h bk = load_bt(kbTc + (size_t)(nt * 16) * CHK, CHK);
        v8f aw = wmma16(a, bk, (v8f){});
        store_ch(&sW[mt * 16][nt * 16], DKV, aw, -1.f);
      }
    }
    __syncthreads();

    // step 4: att = mask(Q_c @ K_c^T) — strict upper zeroed — waves 0..3
    if (wid < 4) {
      int tm = wid & 1, tn = wid >> 1;
      v8f acc = (v8f){};
      for (int kk = 0; kk < DKV; kk += 32) {
        v16h a = load_a(Qc + (size_t)tm * 16 * DKV + kk, DKV);
        v16h bf = load_bt(Kc + (size_t)tn * 16 * DKV + kk, DKV);
        acc = wmma16(a, bf, acc);
      }
      int n_l = lane & 15, mo = (lane >> 4) * 8;
      int Ncol = tn * 16 + n_l;
#pragma unroll
      for (int r = 0; r < 8; ++r) {
        int Mr = tm * 16 + mo + r;
        sAtt[Mr][Ncol] = (_Float16)((Ncol <= Mr) ? acc[r] : 0.f);
      }
    }

    // step 5 (fused): u_adj = u + (-w)@S and o_partial = Q_c@S, sharing each
    // St fragment across two WMMAs.
    v8f oacc[4];
#pragma unroll
    for (int t = 0; t < 4; ++t) oacc[t] = (v8f){};
    for (int kk = 0; kk < DKV; kk += 32) {
      v16h aw = load_a(&sW[mt * 16][kk], DKV);
      v16h aq = load_a(Qc + (size_t)mt * 16 * DKV + kk, DKV);
#pragma unroll
      for (int t = 0; t < 4; ++t) {
        v16h bs = load_bt(St + (size_t)((nt0 + t) * 16) * DKV + kk, DKV);
        ureg[t] = wmma16(aw, bs, ureg[t]);   // u_adj accum
        oacc[t] = wmma16(aq, bs, oacc[t]);   // q@S accum
      }
    }
    __syncthreads();   // all waves done reading -w and (old) attention scratch

    // publish u_adj transposed ([dv][c]) with contiguous 16B stores
#pragma unroll
    for (int t = 0; t < 4; ++t)
      store_chT(&sWT[(nt0 + t) * 16][mt * 16], CHK, ureg[t]);
    __syncthreads();

    // step 6: o = q@S + att @ u_adj  -> delta_out (B,L,H,DV)
    {
      v16h aatt = load_a(&sAtt[mt * 16][0], CHK);
#pragma unroll
      for (int t = 0; t < 4; ++t) {
        int nt = nt0 + t;
        v16h bu = load_bt(&sWT[nt * 16][0], CHK);
        v8f acc = wmma16(aatt, bu, oacc[t]);
        float* ob = dout +
            ((size_t)(b * LSEQ + ch * CHK + mt * 16) * Hh + h) * DKV + nt * 16;
        store_cf(ob, Hh * DKV, acc);
      }
    }
    __syncthreads();

    // step 7: St += u_adj^T @ K_c  (St[dv][dk], K=32): 256 tiles, 32/wave
    for (int t = 0; t < 32; ++t) {
      int tile = wid * 32 + t;
      int tm = tile >> 4, tn = tile & 15;      // tm: dv tile, tn: dk tile
      v16h a = load_a(&sWT[tm * 16][0], CHK);          // A = u_adj^T
      v16h bkt = load_bt(kTc + (size_t)(tn * 16) * CHK, CHK);  // B = K_c
      _Float16* st = St + (size_t)tm * 16 * DKV + tn * 16;
      v8f c = load_ch(st, DKV);
      c = wmma16(a, bkt, c);
      store_ch(st, DKV, c, 1.f);
    }
    __threadfence();
    __syncthreads();
  }
}

// ---------------- gating + softmax + RMSNorm combine -----------------------

__global__ __launch_bounds__(NT) void combine_kernel(
    const float* __restrict__ logits, const float* __restrict__ ltau,
    const float* __restrict__ pval,
    const float* __restrict__ p0, const float* __restrict__ p1,
    const float* __restrict__ p2, const float* __restrict__ p3,
    const float* __restrict__ p4, const float* __restrict__ vdir,
    const float* __restrict__ onw, _Float16* __restrict__ onorm) {
  __shared__ float probs[20];
  __shared__ float pva[4], cms[4], hs[4];
  int row = blockIdx.x, tid = threadIdx.x;
  if (tid < 4) {
    int h = tid;
    float tau = fmaxf(expf(ltau[h]), 0.5f);
    float x[5], mx = -1e30f;
#pragma unroll
    for (int i = 0; i < 5; ++i) {
      x[i] = logits[(size_t)row * 20 + h * 5 + i] / tau;
      mx = fmaxf(mx, x[i]);
    }
    float se = 0.f;
#pragma unroll
    for (int i = 0; i < 5; ++i) { x[i] = expf(x[i] - mx); se += x[i]; }
#pragma unroll
    for (int i = 0; i < 5; ++i) probs[h * 5 + i] = x[i] / se;
    float pv = pval[(size_t)row * 4 + h];
    float fl = 0.01f + 0.09f * (1.f - pv);
    float pa = (1.f - fl) * pv;
    pva[h] = pa;
    cms[h] = 1.f - pa;
    hs[h] = 0.f;
  }
  __syncthreads();
  float oreg[4];
  const float* P[5] = {p0, p1, p2, p3, p4};
#pragma unroll
  for (int j = 0; j < 4; ++j) {      // j == head, channel d == tid
    size_t ix = (size_t)row * Dmodel + j * DKV + tid;
    float ctx = 0.f;
#pragma unroll
    for (int i = 0; i < 5; ++i) ctx += probs[j * 5 + i] * P[i][ix];
    float o = cms[j] * ctx + pva[j] * vdir[ix];
    oreg[j] = o;
    atomicAdd(&hs[j], o * o);
  }
  __syncthreads();
#pragma unroll
  for (int j = 0; j < 4; ++j) {
    float rms = rsqrtf(hs[j] / (float)DKV + 1e-5f);
    onorm[(size_t)row * Dmodel + j * DKV + tid] =
        (_Float16)(oreg[j] * rms * onw[tid]);
  }
}

// ---------------------------------------------------------------------------

extern "C" void kernel_launch(void* const* d_in, const int* in_sizes, int n_in,
                              void* d_out, int out_size, void* d_ws,
                              size_t ws_size, hipStream_t stream) {
  (void)in_sizes; (void)n_in; (void)out_size; (void)ws_size;
  const float* hidden = (const float*)d_in[0];
  const float* Wq   = (const float*)d_in[1];
  const float* Wk   = (const float*)d_in[2];
  const float* Wv   = (const float*)d_in[3];
  const float* qcw  = (const float*)d_in[4];
  const float* kcw  = (const float*)d_in[5];
  const float* vcw  = (const float*)d_in[6];
  const float* Wb   = (const float*)d_in[7];
  const float* f3   = (const float*)d_in[8];
  const float* f7   = (const float*)d_in[9];
  const float* f15  = (const float*)d_in[10];
  const float* f31  = (const float*)d_in[11];
  const float* rW1  = (const float*)d_in[12];
  const float* rb1  = (const float*)d_in[13];
  const float* rW2  = (const float*)d_in[14];
  const float* rb2  = (const float*)d_in[15];
  const float* igW  = (const float*)d_in[16];
  const float* igb  = (const float*)d_in[17];
  const float* ltau = (const float*)d_in[18];
  const float* onw  = (const float*)d_in[19];
  const float* Wo   = (const float*)d_in[20];
  float* out = (float*)d_out;

  const int M = Mrows;                       // 8192 rows
  char* p = (char*)d_ws;
  auto alloc = [&](size_t bytes) -> char* {
    char* r = p;
    p += (bytes + 255) & ~(size_t)255;
    return r;
  };
  _Float16* hiddenH = (_Float16*)alloc((size_t)M * 1024 * 2);
  _Float16* WqT = (_Float16*)alloc((size_t)1024 * 1024 * 2);
  _Float16* WkT = (_Float16*)alloc((size_t)1024 * 1024 * 2);
  _Float16* WvT = (_Float16*)alloc((size_t)1024 * 1024 * 2);
  float* qpre = (float*)alloc((size_t)M * 1024 * 4);
  float* kpre = (float*)alloc((size_t)M * 1024 * 4);
  float* vpre = (float*)alloc((size_t)M * 1024 * 4);
  float* qv = (float*)alloc((size_t)M * 1024 * 4);
  float* kv = (float*)alloc((size_t)M * 1024 * 4);
  float* vv = (float*)alloc((size_t)M * 1024 * 4);   // v_direct
  float* beta = (float*)alloc((size_t)M * 4 * 4);
  float* pvalb = (float*)alloc((size_t)M * 4 * 4);
  size_t nqh = (size_t)16 * LSEQ * DKV;              // B*H*L*256
  _Float16* qh = (_Float16*)alloc(nqh * 2);
  _Float16* kh = (_Float16*)alloc(nqh * 2);
  _Float16* kbh = (_Float16*)alloc(nqh * 2);
  _Float16* kTt = (_Float16*)alloc(nqh * 2);
  _Float16* vTt = (_Float16*)alloc(nqh * 2);
  _Float16* kbTt = (_Float16*)alloc(nqh * 2);
  _Float16* Sg = (_Float16*)alloc((size_t)16 * DKV * DKV * 2);
  float* ddelta = (float*)alloc((size_t)M * 1024 * 4);
  float* path0 = (float*)alloc((size_t)M * 1024 * 4);
  float* path1 = (float*)alloc((size_t)M * 1024 * 4);
  float* path2 = (float*)alloc((size_t)M * 1024 * 4);
  float* path3 = (float*)alloc((size_t)M * 1024 * 4);
  _Float16* rin = (_Float16*)alloc((size_t)M * 6144 * 2);
  _Float16* W1T = (_Float16*)alloc((size_t)6144 * 2048 * 2);
  float* hmid = (float*)alloc((size_t)M * 2048 * 4);
  float* logits = (float*)alloc((size_t)M * 20 * 4);
  _Float16* onormb = (_Float16*)alloc((size_t)M * 1024 * 2);
  _Float16* WoT = (_Float16*)alloc((size_t)1024 * 1024 * 2);

  auto blks = [](long long n) { return (unsigned)((n + NT - 1) / NT); };

  // fp32 -> fp16 operand conversions (weights transposed for B-side)
  cvt_f16<<<blks((long long)M * 1024), NT, 0, stream>>>(hidden, hiddenH, (long long)M * 1024);
  cvt_f16_T<<<blks(1024 * 1024), NT, 0, stream>>>(Wq, WqT, 1024 * 1024, 1024, 1024);
  cvt_f16_T<<<blks(1024 * 1024), NT, 0, stream>>>(Wk, WkT, 1024 * 1024, 1024, 1024);
  cvt_f16_T<<<blks(1024 * 1024), NT, 0, stream>>>(Wv, WvT, 1024 * 1024, 1024, 1024);

  // q/k/v projections (WMMA, async-DMA pipelined)
  dim3 g64x8(M / 128, 1024 / 128);
  wmma_gemm<<<g64x8, NT, 0, stream>>>(hiddenH, WqT, nullptr, qpre, M, 1024, 1024);
  wmma_gemm<<<g64x8, NT, 0, stream>>>(hiddenH, WkT, nullptr, kpre, M, 1024, 1024);
  wmma_gemm<<<g64x8, NT, 0, stream>>>(hiddenH, WvT, nullptr, vpre, M, 1024, 1024);

  // causal dw-conv4 + SiLU
  long long tot = (long long)M * 1024;
  dwconv4_silu<<<blks(tot), NT, 0, stream>>>(qpre, qcw, qv, tot, LSEQ, 1024);
  dwconv4_silu<<<blks(tot), NT, 0, stream>>>(kpre, kcw, kv, tot, LSEQ, 1024);
  dwconv4_silu<<<blks(tot), NT, 0, stream>>>(vpre, vcw, vv, tot, LSEQ, 1024);

  // beta / p_value gates (sigmoid)
  dot_gemm<<<(M * 4 + 7) / 8, NT, 0, stream>>>(hidden, Wb, nullptr, beta, M, 4, 1024, 1);
  dot_gemm<<<(M * 4 + 7) / 8, NT, 0, stream>>>(hidden, igW, igb, pvalb, M, 4, 1024, 1);

  // delta-rule prep + chunkwise scan
  delta_prep<<<M * Hh, NT, 0, stream>>>(qv, kv, vv, beta, qh, kh, kbh, kTt, vTt, kbTt);
  delta_kernel<<<16, NT, 0, stream>>>(qh, kh, kbh, kTt, vTt, kbTt, Sg, ddelta);

  // FIR paths (also pack f16 router-input columns)
  fir_kernel<<<blks(tot), NT, 0, stream>>>(vv, f3, path0, rin, 3, 1024, tot, LSEQ, 1024);
  fir_kernel<<<blks(tot), NT, 0, stream>>>(vv, f7, path1, rin, 7, 2048, tot, LSEQ, 1024);
  fir_kernel<<<blks(tot), NT, 0, stream>>>(vv, f15, path2, rin, 15, 3072, tot, LSEQ, 1024);
  fir_kernel<<<blks(tot), NT, 0, stream>>>(vv, f31, path3, rin, 31, 4096, tot, LSEQ, 1024);
  pack16<<<blks(tot), NT, 0, stream>>>(hidden, rin, tot, 1024, 6144, 0);
  pack16<<<blks(tot), NT, 0, stream>>>(ddelta, rin, tot, 1024, 6144, 5120);

  // router MLP: hmid = gelu(rin @ W1 + b1); logits = hmid @ W2 + b2
  cvt_f16_T<<<blks((long long)6144 * 2048), NT, 0, stream>>>(rW1, W1T,
                                                             (long long)6144 * 2048, 2048, 6144);
  dim3 gR(M / 128, 2048 / 128);
  wmma_gemm<<<gR, NT, 0, stream>>>(rin, W1T, rb1, hmid, M, 2048, 6144);
  gelu_kernel<<<blks((long long)M * 2048), NT, 0, stream>>>(hmid, (long long)M * 2048);
  dot_gemm<<<(M * 20 + 7) / 8, NT, 0, stream>>>(hmid, rW2, rb2, logits, M, 20, 2048, 0);

  // combine paths, gate, RMSNorm
  combine_kernel<<<M, NT, 0, stream>>>(logits, ltau, pvalb, path0, path1, path2,
                                       path3, ddelta, vv, onw, onormb);

  // output projection
  cvt_f16_T<<<blks(1024 * 1024), NT, 0, stream>>>(Wo, WoT, 1024 * 1024, 1024, 1024);
  wmma_gemm<<<g64x8, NT, 0, stream>>>(onormb, WoT, nullptr, out, M, 1024, 1024);
}
